// Decoder_47201690583385
// MI455X (gfx1250) — matrix operand order repaired and checked by reading, not compiled, not run on hardware
//
#include <hip/hip_runtime.h>
#include <hip/hip_bf16.h>

// GRU decode for B=64, S=512, H=512, IN=1024 on gfx1250 (MI455X).
// Phase 1: gi = (x*mask) @ W_ih^T + b_ih  via v_wmma_f32_16x16x32_bf16 (f32 accum).
// Phase 2: persistent scan kernel, one device-wide barrier per timestep,
//          per-wave register-resident r/z/n accumulators.

typedef __attribute__((ext_vector_type(16))) __bf16 v16bf;
typedef __attribute__((ext_vector_type(8)))  float  v8f;

#define B_   64
#define S_   512
#define H_   512
#define IN_  1024
#define TG_  1536   // 3*H

// ---------------- workspace layout (bytes, all 256-aligned) ----------------
#define OFF_WIH  ((size_t)0)                         // 1536*1024 bf16 = 3 MB
#define OFF_WHH  ((size_t)3145728)                   // 1536*512  bf16 = 1.5 MB
#define OFF_GI   ((size_t)4718592)                   // 32768*1536 f32 = 192 MB
#define OFF_HA   ((size_t)206045184)                 // 64*512 f32
#define OFF_HB   ((size_t)206176256)                 // 64*512 f32
#define OFF_CNT  ((size_t)206307328)                 // barrier counter

// ---------------- WMMA fragment helpers (contiguous K per lane half, same order for A and B) ----------------
// A 16x32 bf16 (MxK): lane&15 = row M; per lane, elems 0..15 hold
// K = klo..klo+15, klo = (lane>=16)*16.
static __device__ __forceinline__ v16bf load_a_f32(const float* __restrict__ base,
                                                   int ld, int row0, int k0, float scale) {
  const int lane = threadIdx.x & 31;
  const int r    = row0 + (lane & 15);
  const int klo  = k0 + ((lane >> 4) << 4);
  const float* p = base + (size_t)r * ld + klo;
  v16bf a;
#pragma unroll
  for (int i = 0; i < 8; ++i) {
    a[i]     = (__bf16)(p[i]      * scale);
    a[i + 8] = (__bf16)(p[i + 8] * scale);
  }
  return a;
}

// B 32x16 bf16 (KxN): lane&15 = col N; per lane a contiguous run of 16 K
// values starting at (lane>=16)*16.  Source is row-major [N,K] (i.e. W[n,k]).
static __device__ __forceinline__ v16bf load_b_bf16(const __bf16* __restrict__ base,
                                                    int ld, int col0, int k0) {
  const int lane = threadIdx.x & 31;
  const int n    = col0 + (lane & 15);
  const int koff = k0 + ((lane >> 4) << 4);
  const __bf16* p = base + (size_t)n * ld + koff;
  v16bf b;
#pragma unroll
  for (int i = 0; i < 16; ++i) b[i] = p[i];
  return b;
}

static __device__ __forceinline__ v8f wmma_bf16(v16bf a, v16bf b, v8f c) {
  // (neg_a, A, neg_b, B, c_mod, C, reuse_a, reuse_b)
  return __builtin_amdgcn_wmma_f32_16x16x32_bf16(false, a, false, b, (short)0, c,
                                                 false, false);
}

static __device__ __forceinline__ float sigmoidf_(float x) {
  return 1.0f / (1.0f + __expf(-x));
}

// ---------------- Phase 0: weight conversion + state init ----------------
__global__ void prep_kernel(const float* __restrict__ Wih, const float* __restrict__ Whh,
                            const float* __restrict__ h0,
                            __bf16* __restrict__ WihB, __bf16* __restrict__ WhhB,
                            float* __restrict__ hbuf, unsigned* __restrict__ counter) {
  const size_t tid    = (size_t)blockIdx.x * blockDim.x + threadIdx.x;
  const size_t stride = (size_t)gridDim.x * blockDim.x;
  for (size_t t = tid; t < (size_t)TG_ * IN_; t += stride) WihB[t] = (__bf16)Wih[t];
  for (size_t t = tid; t < (size_t)TG_ * H_;  t += stride) WhhB[t] = (__bf16)Whh[t];
  for (size_t t = tid; t < (size_t)B_ * H_;   t += stride) hbuf[t] = h0[t];
  if (tid == 0) *counter = 0u;   // re-armed every launch -> graph-replay safe
}

// ---------------- Phase 1: gi = (x*mask) @ W_ih^T + b_ih ----------------
// Grid: (TG_/64, (B_*S_)/128), block 256 = 8 waves.  Wave w owns rows
// [m0+16w, m0+16w+16) x cols [n0, n0+64): 4 accumulators, 32 K-steps.
__global__ void __launch_bounds__(256) gi_gemm_kernel(const float* __restrict__ x,
                                                      const float* __restrict__ masks,
                                                      const __bf16* __restrict__ WihB,
                                                      const float* __restrict__ b_ih,
                                                      float* __restrict__ gi) {
  const int lane = threadIdx.x & 31;
  const int wave = threadIdx.x >> 5;                 // 0..7
  const int m0   = blockIdx.y * 128 + wave * 16;     // row base for this wave
  const int n0   = blockIdx.x * 64;                  // col base for this WG
  const float mk = masks[m0 + (lane & 15)];          // masks flat [B*S]

  v8f acc[4] = {v8f{}, v8f{}, v8f{}, v8f{}};
  for (int k = 0; k < IN_; k += 32) {
    const v16bf a = load_a_f32(x, IN_, m0, k, mk);
#pragma unroll
    for (int sub = 0; sub < 4; ++sub) {
      const v16bf b = load_b_bf16(WihB, IN_, n0 + sub * 16, k);
      acc[sub] = wmma_bf16(a, b, acc[sub]);
    }
  }
  // Epilogue: D elem e sits at (M = 8*(lane>=16)+e, N = lane&15); add bias.
#pragma unroll
  for (int sub = 0; sub < 4; ++sub) {
    const int n = n0 + sub * 16 + (lane & 15);
    const float bias = b_ih[n];
#pragma unroll
    for (int e = 0; e < 8; ++e) {
      const int m = m0 + ((lane >> 4) << 3) + e;
      gi[(size_t)m * TG_ + n] = acc[sub][e] + bias;
    }
  }
}

// ---------------- Phase 2: persistent GRU scan ----------------
// 32 WGs x 128 threads (4 waves).  WG g owns H-columns [16g, 16g+16);
// wave w owns batch rows [16w, 16w+16).  Each wave holds the r/z/n 16x16
// tiles in registers, so the gate math needs no LDS and no intra-WG sync.
// One atomic grid barrier per step publishes the double-buffered h.
__global__ void __launch_bounds__(128) gru_scan_kernel(const float* __restrict__ gi,
                                                       const __bf16* __restrict__ WhhB,
                                                       const float* __restrict__ b_hh,
                                                       const float* __restrict__ masks,
                                                       float* __restrict__ hA,
                                                       float* __restrict__ hB,
                                                       float* __restrict__ out,
                                                       unsigned* __restrict__ counter) {
  const int lane  = threadIdx.x & 31;
  const int wave  = threadIdx.x >> 5;     // 0..3
  const int j0    = blockIdx.x * 16;      // column slice of H
  const int m0    = wave * 16;            // batch-row base
  const int j     = j0 + (lane & 15);
  const float bhr = b_hh[j];
  const float bhz = b_hh[H_ + j];
  const float bhn = b_hh[2 * H_ + j];
  const unsigned nwg = gridDim.x;

  for (int s = 0; s < S_; ++s) {
    const float* hin  = (s & 1) ? hB : hA;
    float*       hout = (s & 1) ? hA : hB;

    // Warm next step's gi rows into cache while we do this step's GEMM.
    if (s + 1 < S_) {
      const size_t pr = ((size_t)(m0 + ((lane >> 4) << 3)) * S_ + (s + 1)) * TG_ + j;
      __builtin_prefetch(gi + pr, 0, 1);
    }

    v8f accR = {}, accZ = {}, accN = {};
    for (int k = 0; k < H_; k += 32) {
      const v16bf a  = load_a_f32(hin, H_, m0, k, 1.0f);
      const v16bf bR = load_b_bf16(WhhB, H_, j0,          k);
      const v16bf bZ = load_b_bf16(WhhB, H_, H_ + j0,     k);
      const v16bf bN = load_b_bf16(WhhB, H_, 2 * H_ + j0, k);
      accR = wmma_bf16(a, bR, accR);
      accZ = wmma_bf16(a, bZ, accZ);
      accN = wmma_bf16(a, bN, accN);
    }

    // Gate math, fully register-resident per wave.
#pragma unroll
    for (int e = 0; e < 8; ++e) {
      const int b      = m0 + ((lane >> 4) << 3) + e;
      const size_t row = (size_t)b * S_ + s;        // flattened (b, s)
      const float* gp  = gi + row * TG_;
      const float ir = gp[j];
      const float iz = gp[H_ + j];
      const float inn = gp[2 * H_ + j];
      const float r  = sigmoidf_(ir + accR[e] + bhr);
      const float z  = sigmoidf_(iz + accZ[e] + bhz);
      const float n  = tanhf(inn + r * (accN[e] + bhn));
      const float hp = hin[b * H_ + j];
      const float hn = (1.0f - z) * n + z * hp;     // torch GRUCell update
      hout[b * H_ + j] = hn;
      out[row * H_ + j] = hn * masks[row];
    }

    // ---- device-wide barrier: publish hout before anyone reads it ----
    __threadfence();
    __syncthreads();
    if (threadIdx.x == 0) {
      const unsigned target = (unsigned)(s + 1) * nwg;
      atomicAdd(counter, 1u);
      while (__hip_atomic_load(counter, __ATOMIC_ACQUIRE,
                               __HIP_MEMORY_SCOPE_AGENT) < target) {
        __builtin_amdgcn_s_sleep(2);
      }
    }
    __syncthreads();
    __threadfence();
  }
}

// ---------------- host-side launch ----------------
extern "C" void kernel_launch(void* const* d_in, const int* in_sizes, int n_in,
                              void* d_out, int out_size, void* d_ws, size_t ws_size,
                              hipStream_t stream) {
  (void)in_sizes; (void)n_in; (void)out_size; (void)ws_size;
  const float* inputs      = (const float*)d_in[0];   // [B,S,2H] f32
  const float* init_states = (const float*)d_in[1];   // [B,H]    f32
  const float* masks       = (const float*)d_in[2];   // [B,S]    f32
  /* d_in[3] = d_in indices (int64) -- unused by the math */
  const float* W_ih        = (const float*)d_in[4];   // [3H,2H]  f32
  const float* W_hh        = (const float*)d_in[5];   // [3H,H]   f32
  const float* b_ih        = (const float*)d_in[6];   // [3H]     f32
  const float* b_hh        = (const float*)d_in[7];   // [3H]     f32
  float* out = (float*)d_out;                          // [B,S,H]  f32

  char* ws = (char*)d_ws;
  __bf16*   WihB    = (__bf16*)(ws + OFF_WIH);
  __bf16*   WhhB    = (__bf16*)(ws + OFF_WHH);
  float*    gi      = (float*)(ws + OFF_GI);
  float*    hA      = (float*)(ws + OFF_HA);
  float*    hB      = (float*)(ws + OFF_HB);
  unsigned* counter = (unsigned*)(ws + OFF_CNT);

  prep_kernel<<<2048, 256, 0, stream>>>(W_ih, W_hh, init_states,
                                        WihB, WhhB, hA, counter);

  dim3 gGrid(TG_ / 64, (B_ * S_) / 128);
  gi_gemm_kernel<<<gGrid, 256, 0, stream>>>(inputs, masks, WihB, b_ih, gi);

  gru_scan_kernel<<<H_ / 16, 128, 0, stream>>>(gi, WhhB, b_hh, masks,
                                               hA, hB, out, counter);
}
